// MutilQueryAttention_63118839382028
// MI455X (gfx1250) — matrix operand order repaired and checked by reading, not compiled, not run on hardware
//
#include <hip/hip_runtime.h>
#include <hip/hip_bf16.h>
#include <cmath>

// ---------------------------------------------------------------------------
// MQA forward for MI455X (gfx1250, wave32, WMMA + double-buffered
// async-to-LDS pipelining).
// B=2, S=2048, D=2048, H=16, HD=128 (single shared KV head).
// ---------------------------------------------------------------------------

typedef __attribute__((ext_vector_type(16))) __bf16 v16bf;
typedef __attribute__((ext_vector_type(8)))  __bf16 v8bf;
typedef __attribute__((ext_vector_type(8)))  float  v8f;

#define BB    2
#define SS    2048
#define DD    2048
#define HH    16
#define HDIM  128
#define SCALE 0.08838834764831845f   // 1/sqrt(128)

__device__ __forceinline__ v8f wmma_bf(v16bf a, v16bf b, v8f c) {
  return __builtin_amdgcn_wmma_f32_16x16x32_bf16(false, a, false, b,
                                                 (short)0, c, false, false);
}

// CDNA5 async copy: 16B global -> LDS per lane, tracked by ASYNCcnt.
// GV mode: 64-bit vaddr in VGPR pair, LDS byte offset in VDST, saddr = off.
__device__ __forceinline__ void async_load16(unsigned lds_off, const void* gaddr) {
  asm volatile("global_load_async_to_lds_b128 %0, %1, off"
               :: "v"(lds_off), "v"((unsigned long long)(uintptr_t)gaddr)
               : "memory");
}
// Wait until ASYNCcnt <= N. Async loads complete in order, so in a
// ping-pong pipeline waiting for <= (chunks just issued) guarantees the
// previous buffer's chunks have all landed.
template <int N>
__device__ __forceinline__ void wait_async() {
  asm volatile("s_wait_asynccnt %0" :: "i"(N) : "memory");
}

// A-fragment (16x32 bf16): lane L -> row = L%16, kbase = (L/16)*16;
// elements 0..15 = k kbase..+15 (contiguous).
__device__ __forceinline__ v16bf make_afrag(v8bf lo, v8bf hi8) {
  v16bf r;
#pragma unroll
  for (int i = 0; i < 8; ++i) { r[i] = lo[i]; r[8 + i] = hi8[i]; }
  return r;
}
__device__ __forceinline__ v16bf load_afrag_g(const __bf16* rowptr, int kbase) {
  return make_afrag(*(const v8bf*)(rowptr + kbase),
                    *(const v8bf*)(rowptr + kbase + 8));
}

// ------------------------- conversion kernels ------------------------------

__global__ void cvt_bf16(__bf16* __restrict__ dst, const float* __restrict__ src, int n) {
  int i = blockIdx.x * blockDim.x + threadIdx.x;
  if (i < n) dst[i] = (__bf16)src[i];
}

// src [K,N] f32 row-major -> dst [N,K] bf16 row-major (transposed weights)
__global__ void cvt_t_bf16(__bf16* __restrict__ dst, const float* __restrict__ src,
                           int K, int N) {
  int i = blockIdx.x * blockDim.x + threadIdx.x;   // i = n*K + k
  if (i < K * N) {
    int k = i % K;
    int n = i / K;
    dst[i] = (__bf16)src[(size_t)k * N + n];
  }
}

// V [B,S,HD] bf16 -> Vt [B,HD,S] bf16
__global__ void transpose_v(__bf16* __restrict__ dst, const __bf16* __restrict__ src) {
  int i = blockIdx.x * blockDim.x + threadIdx.x;   // i = ((b*HD)+d)*S + s
  if (i < BB * HDIM * SS) {
    int s = i % SS;
    int d = (i / SS) % HDIM;
    int b = i / (SS * HDIM);
    dst[i] = src[((size_t)(b * SS + s)) * HDIM + d];
  }
}

// ------------------------- WMMA GEMM (TN) ----------------------------------
// C[M,N] = A[M,K] * B[K,N] with B given transposed (Bt[N,K]).
// Block = 128 threads = 4 waves computing a 64x64 C tile.
// Double-buffered LDS: async copies for K-step i+1 are issued before the
// WMMAs of step i, hiding global->LDS latency behind the matrix pipe.
template <int OUTF32>
__global__ __launch_bounds__(128)
void gemm_tn(const __bf16* __restrict__ A, const __bf16* __restrict__ Bt,
             float* __restrict__ Cf, __bf16* __restrict__ Cb,
             int M, int N, int K) {
  __shared__ __bf16 Asm[2][64][32];    // 8 KB
  __shared__ __bf16 Bsm[2][64][32];    // 8 KB
  const int tid  = threadIdx.x;
  const int lane = tid & 31;
  const int wave = tid >> 5;
  const int row  = lane & 15;
  const int hi   = lane >> 4;
  const int m0   = blockIdx.y * 64;
  const int n0   = blockIdx.x * 64;

  // 64x32 bf16 = 4KB per tile = 256 16B chunks; 2 chunks/thread/tile,
  // 4 async ops per thread per K-step.
  auto issue = [&](int buf, int kk) {
#pragma unroll
    for (int i = 0; i < 2; ++i) {
      int id = i * 128 + tid;          // 0..255
      int r  = id >> 2;                // 0..63
      int c8 = (id & 3) * 8;           // 0,8,16,24
      async_load16((unsigned)(uintptr_t)&Asm[buf][r][c8],
                   A + (size_t)(m0 + r) * K + kk + c8);
      async_load16((unsigned)(uintptr_t)&Bsm[buf][r][c8],
                   Bt + (size_t)(n0 + r) * K + kk + c8);
    }
  };

  v8f acc[4];
#pragma unroll
  for (int j = 0; j < 4; ++j) acc[j] = (v8f){};

  issue(0, 0);
  for (int kk = 0; kk < K; kk += 32) {
    const int buf = (kk >> 5) & 1;
    if (kk + 32 < K) {
      issue(buf ^ 1, kk + 32);
      wait_async<4>();                 // previous buffer fully landed
    } else {
      wait_async<0>();
    }
    __syncthreads();

    v16bf af = make_afrag(*(const v8bf*)&Asm[buf][wave * 16 + row][hi * 16],
                          *(const v8bf*)&Asm[buf][wave * 16 + row][hi * 16 + 8]);
#pragma unroll
    for (int j = 0; j < 4; ++j) {
      v16bf bm = *(const v16bf*)&Bsm[buf][j * 16 + row][hi * 16];
      acc[j] = wmma_bf(af, bm, acc[j]);
    }
    __syncthreads();   // all readers done before this buf is refilled at kk+64
  }

  // C/D layout: element r -> row (tile + r + 8*hi), col (tile + lane&15)
  const int orow0 = m0 + wave * 16 + 8 * hi;
#pragma unroll
  for (int j = 0; j < 4; ++j) {
    const int ocol = n0 + j * 16 + row;
#pragma unroll
    for (int r = 0; r < 8; ++r) {
      if (OUTF32) Cf[(size_t)(orow0 + r) * N + ocol] = acc[j][r];
      else        Cb[(size_t)(orow0 + r) * N + ocol] = (__bf16)acc[j][r];
    }
  }
}

// ------------------------- flash MQA attention -----------------------------
// Block = 4 waves, each owning 16 query rows of the same (b,h); the K/V
// tiles for each 32-key block are shared by all waves (MQA) and staged via
// double-buffered async copies overlapping the WMMA/softmax of the current
// block.
__global__ __launch_bounds__(128)
void mqa_attn(const __bf16* __restrict__ Q, const __bf16* __restrict__ K,
              const __bf16* __restrict__ Vt, __bf16* __restrict__ HO) {
  __shared__ __bf16 Ksm[2][32][HDIM];  // 16 KB: keys x d
  __shared__ __bf16 Vsm[2][HDIM][32];  // 16 KB: d x keys
  __shared__ __bf16 Plds[4][16][32];   //  4 KB: per-wave P bounce

  const int tid  = threadIdx.x;
  const int lane = tid & 31;
  const int wave = tid >> 5;
  const int row  = lane & 15;
  const int hi   = lane >> 4;
  const int qt   = blockIdx.x * 4 + wave;
  const int h    = blockIdx.y;
  const int b    = blockIdx.z;
  const int q0   = qt * 16;

  const __bf16* Qrow  = Q  + ((size_t)(b * SS + q0 + row)) * (HH * HDIM) + h * HDIM;
  const __bf16* Kbase = K  + (size_t)b * SS * HDIM;
  const __bf16* Vbase = Vt + (size_t)b * HDIM * SS;

  // stage K[32][128] and Vt[128][32]: 512 chunks each, 8 async ops/thread
  auto issue = [&](int buf, int key0) {
#pragma unroll
    for (int i = 0; i < 4; ++i) {
      int id  = i * 128 + tid;         // 0..511
      int kr  = id >> 4;               // 0..31
      int kc8 = (id & 15) * 8;         // 0..120
      async_load16((unsigned)(uintptr_t)&Ksm[buf][kr][kc8],
                   Kbase + (size_t)(key0 + kr) * HDIM + kc8);
      int vd  = id >> 2;               // 0..127
      int vc8 = (id & 3) * 8;          // 0,8,16,24
      async_load16((unsigned)(uintptr_t)&Vsm[buf][vd][vc8],
                   Vbase + (size_t)vd * SS + key0 + vc8);
    }
  };

  v16bf qf[4];
#pragma unroll
  for (int kc = 0; kc < 4; ++kc) qf[kc] = load_afrag_g(Qrow, kc * 32 + hi * 16);

  v8f acco[8];
#pragma unroll
  for (int g = 0; g < 8; ++g) acco[g] = (v8f){};
  float m[8], l[8];
#pragma unroll
  for (int r = 0; r < 8; ++r) { m[r] = -INFINITY; l[r] = 0.0f; }

  issue(0, 0);
  for (int key0 = 0; key0 < SS; key0 += 32) {
    const int buf = (key0 >> 5) & 1;
    if (key0 + 32 < SS) {
      issue(buf ^ 1, key0 + 32);
      wait_async<8>();                 // previous buffer fully landed
    } else {
      wait_async<0>();
    }
    __syncthreads();

    // scores: S0 = keys [key0,key0+16), S1 = keys [key0+16,key0+32)
    v8f s0 = {}, s1 = {};
#pragma unroll
    for (int kc = 0; kc < 4; ++kc) {
      v16bf k0 = *(const v16bf*)&Ksm[buf][row][kc * 32 + hi * 16];
      v16bf k1 = *(const v16bf*)&Ksm[buf][16 + row][kc * 32 + hi * 16];
      s0 = wmma_bf(qf[kc], k0, s0);
      s1 = wmma_bf(qf[kc], k1, s1);
    }

    // online softmax; element r lives on row (r + 8*hi), cols = lanes of half
    float p0[8], p1[8];
#pragma unroll
    for (int r = 0; r < 8; ++r) {
      float a0 = s0[r] * SCALE;
      float a1 = s1[r] * SCALE;
      float mx = fmaxf(a0, a1);
#pragma unroll
      for (int off = 1; off < 16; off <<= 1) mx = fmaxf(mx, __shfl_xor(mx, off, 16));
      float mn = fmaxf(m[r], mx);
      float c  = __expf(m[r] - mn);
      p0[r] = __expf(a0 - mn);
      p1[r] = __expf(a1 - mn);
      float rs = p0[r] + p1[r];
#pragma unroll
      for (int off = 1; off < 16; off <<= 1) rs += __shfl_xor(rs, off, 16);
      l[r] = l[r] * c + rs;
      m[r] = mn;
#pragma unroll
      for (int g = 0; g < 8; ++g) acco[g][r] = acco[g][r] * c;
    }

    // P: C-layout -> A-layout through LDS (wave-private buffer)
#pragma unroll
    for (int r = 0; r < 8; ++r) {
      Plds[wave][8 * hi + r][row]      = (__bf16)p0[r];
      Plds[wave][8 * hi + r][16 + row] = (__bf16)p1[r];
    }
    __builtin_amdgcn_wave_barrier();
    v16bf pf;
#pragma unroll
    for (int j = 0; j < 8; ++j) {
      pf[j]     = Plds[wave][row][hi * 16 + j];
      pf[8 + j] = Plds[wave][row][hi * 16 + 8 + j];
    }
    __builtin_amdgcn_wave_barrier();

    // O += P (16x32) x V (32x16) per 16-wide d-group
#pragma unroll
    for (int g = 0; g < 8; ++g) {
      v16bf vf = *(const v16bf*)&Vsm[buf][g * 16 + row][hi * 16];
      acco[g] = wmma_bf(pf, vf, acco[g]);
    }
    __syncthreads();   // all readers done before this buf is refilled
  }

  // normalize + store head_out [B,S,H*HD] bf16
#pragma unroll
  for (int g = 0; g < 8; ++g) {
#pragma unroll
    for (int r = 0; r < 8; ++r) {
      float o = acco[g][r] / l[r];
      HO[((size_t)(b * SS + q0 + 8 * hi + r)) * (HH * HDIM) + h * HDIM + g * 16 + row] =
          (__bf16)o;
    }
  }
}

// ------------------------------ launch -------------------------------------

extern "C" void kernel_launch(void* const* d_in, const int* in_sizes, int n_in,
                              void* d_out, int out_size, void* d_ws, size_t ws_size,
                              hipStream_t stream) {
  (void)in_sizes; (void)n_in; (void)out_size; (void)ws_size;
  const float* Xin = (const float*)d_in[0];
  const float* Wq  = (const float*)d_in[1];
  const float* Wk  = (const float*)d_in[2];
  const float* Wv  = (const float*)d_in[3];
  const float* Wo  = (const float*)d_in[4];
  float* out = (float*)d_out;

  char* ws = (char*)d_ws;
  size_t off = 0;
  auto carve = [&](size_t elems) {
    __bf16* p = (__bf16*)(ws + off);
    off += ((elems * sizeof(__bf16)) + 255) & ~(size_t)255;
    return p;
  };
  __bf16* Xbf = carve((size_t)BB * SS * DD);      // 16 MB
  __bf16* Wqt = carve((size_t)DD * DD);           //  8 MB
  __bf16* Wkt = carve((size_t)HDIM * DD);         // .5 MB
  __bf16* Wvt = carve((size_t)HDIM * DD);         // .5 MB
  __bf16* Wot = carve((size_t)DD * DD);           //  8 MB
  __bf16* Qb  = carve((size_t)BB * SS * DD);      // 16 MB
  __bf16* Kb  = carve((size_t)BB * SS * HDIM);    //  1 MB
  __bf16* Vb  = carve((size_t)BB * SS * HDIM);    //  1 MB
  __bf16* Vtb = carve((size_t)BB * SS * HDIM);    //  1 MB
  __bf16* HOb = carve((size_t)BB * SS * DD);      // 16 MB

  int n = BB * SS * DD;
  cvt_bf16<<<(n + 255) / 256, 256, 0, stream>>>(Xbf, Xin, n);
  cvt_t_bf16<<<(DD * DD + 255) / 256, 256, 0, stream>>>(Wqt, Wq, DD, DD);
  cvt_t_bf16<<<(DD * HDIM + 255) / 256, 256, 0, stream>>>(Wkt, Wk, DD, HDIM);
  cvt_t_bf16<<<(DD * HDIM + 255) / 256, 256, 0, stream>>>(Wvt, Wv, DD, HDIM);
  cvt_t_bf16<<<(DD * DD + 255) / 256, 256, 0, stream>>>(Wot, Wo, DD, DD);

  dim3 blk(128);
  // Q = X @ Wq   (M=4096, N=2048, K=2048), 64x64 C tiles
  gemm_tn<0><<<dim3(DD / 64, BB * SS / 64), blk, 0, stream>>>(
      Xbf, Wqt, nullptr, Qb, BB * SS, DD, DD);
  // K = X @ Wk   (N=128)
  gemm_tn<0><<<dim3(HDIM / 64, BB * SS / 64), blk, 0, stream>>>(
      Xbf, Wkt, nullptr, Kb, BB * SS, HDIM, DD);
  // V = X @ Wv
  gemm_tn<0><<<dim3(HDIM / 64, BB * SS / 64), blk, 0, stream>>>(
      Xbf, Wvt, nullptr, Vb, BB * SS, HDIM, DD);

  n = BB * HDIM * SS;
  transpose_v<<<(n + 255) / 256, 256, 0, stream>>>(Vtb, Vb);

  // flash attention: grid (S/64 qtiles-of-4, H, B)
  mqa_attn<<<dim3(SS / 64, HH, BB), blk, 0, stream>>>(Qb, Kb, Vtb, HOb);

  // out = HO @ Wo  (f32 output)
  gemm_tn<1><<<dim3(DD / 64, BB * SS / 64), blk, 0, stream>>>(
      HOb, Wot, out, nullptr, BB * SS, DD, DD);
}
